// AttentionNet_54932631715876
// MI455X (gfx1250) — compile-verified, matrix operand order checked by reading, not run on hardware
//
#include <hip/hip_runtime.h>

typedef __attribute__((ext_vector_type(16))) _Float16 v16h;
typedef __attribute__((ext_vector_type(8)))  float    v8f;

#define LDSTR 65
#define EPSF  1e-6f
#define LOG2E 1.44269504088896340736f

__device__ __forceinline__ float fast_sigmoid(float t) {
    // sigmoid(t) = 1 / (1 + exp(-t)) via v_exp_f32 (exp2) + v_rcp_f32
    float e = __builtin_amdgcn_exp2f(t * -LOG2E);
    return __builtin_amdgcn_rcpf(1.0f + e);
}

// One AttentionVector pass on a 64x64 matrix resident in LDS.
//   x_eff[a][k] = xmat[a*LDSTR+k] * (ROWS ? rs[a] : 1) * (COLS ? cs[k] : 1)
//   c[a,a'] = mean_k sigmoid(W0*x_eff[a',k] + W1*x_eff[a,k] + b)
//   m[a] = relu( (mean_a' c) / (max_a' c + eps) * sharp + th )
//   outvec = softmax(m) [ / (max + eps) unless LAST ]
// The mean_k is performed on the WMMA pipe: A = sigmoid tile (f16),
// B = constant 1/64, accumulated in f32.
template <bool ROWS, bool COLS, bool LAST>
__device__ void att_phase(const float* __restrict__ xmat,   // LDS [64][LDSTR]
                          const float* __restrict__ rs,     // row scale (LDS) or nullptr
                          const float* __restrict__ cs,     // col scale (LDS) or nullptr
                          float W0, float W1, float bb,
                          float sharp, float th,
                          float* __restrict__ mvals,        // LDS[64] scratch
                          float* __restrict__ outvec)       // LDS[64] or global row
{
    const int tid  = threadIdx.x;
    const int lane = tid & 31;
    const int wave = tid >> 5;
    const int hi   = (lane >= 16) ? 1 : 0;
    const int mrow = lane & 15;

    v16h bfrag;
#pragma unroll
    for (int e = 0; e < 16; ++e) bfrag[e] = (_Float16)0.015625f; // 1/64: B matrix of the mean

    for (int t = 0; t < 8; ++t) {
        const int a = wave * 8 + t;                 // output row this wave computes now
        const float rsa = ROWS ? rs[a] : 1.0f;
        const float wi  = W1 * rsa;
        const float* xa = xmat + a * LDSTR;

        v8f acc[4];
#pragma unroll
        for (int b = 0; b < 4; ++b)
#pragma unroll
            for (int r = 0; r < 8; ++r) acc[b][r] = 0.0f;

#pragma unroll
        for (int ch = 0; ch < 2; ++ch) {
            const int kbase = ch * 32 + hi * 8;
            // Per-chunk precompute: base[e] = wi*x_eff[a,k]+b and col factor,
            // reused across all 4 a'-blocks (k set is identical).
            float base[16], csk[16];
#pragma unroll
            for (int e = 0; e < 16; ++e) {
                const int k = kbase + (e & 7) + ((e >> 3) << 4); // ISA 16-bit A layout
                const float c = COLS ? cs[k] : 1.0f;
                csk[e]  = c;
                base[e] = __builtin_fmaf(wi * xa[k], c, bb);
            }
#pragma unroll
            for (int b = 0; b < 4; ++b) {
                const int arow = b * 16 + mrow;      // a' row held by this lane
                const float wj = W0 * (ROWS ? rs[arow] : 1.0f);
                const float* xr = xmat + arow * LDSTR;
                v16h af;
#pragma unroll
                for (int e = 0; e < 16; ++e) {
                    const int k = kbase + (e & 7) + ((e >> 3) << 4);
                    float p   = csk[e] * xr[k];
                    float lin = __builtin_fmaf(wj, p, base[e]);
                    af[e] = (_Float16)fast_sigmoid(lin);
                }
                // D[a'_local, *] += (1/64) * sum_K A[a'_local, K]  (replicated over cols)
                acc[b] = __builtin_amdgcn_wmma_f32_16x16x32_f16(
                    false, af, false, bfrag, (short)0, acc[b], false, false);
            }
        }

        // Row sum & max over all 64 a': lane holds 8 rows per block (lo/hi half),
        // pair with lane^16 to cover all 16 rows of each block.
        float sum = 0.0f, mx = -1e30f;
#pragma unroll
        for (int b = 0; b < 4; ++b)
#pragma unroll
            for (int r = 0; r < 8; ++r) {
                float v = acc[b][r];
                sum += v;
                mx = fmaxf(mx, v);
            }
        sum += __shfl_xor(sum, 16, 32);
        mx   = fmaxf(mx, __shfl_xor(mx, 16, 32));

        float m = (sum * 0.015625f) * __builtin_amdgcn_rcpf(mx + EPSF);
        m = fmaxf(__builtin_fmaf(m, sharp, th), 0.0f);
        if (lane == 0) mvals[a] = m;
    }
    __syncthreads();

    // Softmax over the 64 m-values (wave 0, 2 elements/lane, shuffle reductions)
    if (wave == 0) {
        float v0 = mvals[lane];
        float v1 = mvals[lane + 32];
        float mx = fmaxf(v0, v1);
#pragma unroll
        for (int o = 16; o > 0; o >>= 1) mx = fmaxf(mx, __shfl_xor(mx, o, 32));
        float e0 = __builtin_amdgcn_exp2f((v0 - mx) * LOG2E);
        float e1 = __builtin_amdgcn_exp2f((v1 - mx) * LOG2E);
        float s  = e0 + e1;
#pragma unroll
        for (int o = 16; o > 0; o >>= 1) s += __shfl_xor(s, o, 32);
        float rz = __builtin_amdgcn_rcpf(s);
        float o0 = e0 * rz, o1 = e1 * rz;
        if (!LAST) {
            // max(softmax) == 1/Z == rz (the argmax term is exp(0)=1 exactly)
            float inv = __builtin_amdgcn_rcpf(rz + EPSF);
            o0 *= inv; o1 *= inv;
        }
        outvec[lane]      = o0;
        outvec[lane + 32] = o1;
    }
    __syncthreads();
}

__global__ __launch_bounds__(256) void attnet_kernel(
    const float* __restrict__ x,
    const float* __restrict__ fcW, const float* __restrict__ fcB,
    const float* __restrict__ p1W, const float* __restrict__ p1b,
    const float* __restrict__ p1s, const float* __restrict__ p1t,
    const float* __restrict__ p2W, const float* __restrict__ p2b,
    const float* __restrict__ p2s, const float* __restrict__ p2t,
    const float* __restrict__ oW,  const float* __restrict__ ob,
    const float* __restrict__ oS,  const float* __restrict__ oT,
    float* __restrict__ out)
{
    __shared__ float h0 [64 * LDSTR];  // channel 0, row-major [i][j]
    __shared__ float h1t[64 * LDSTR];  // channel 1 transposed  [j][i]
    __shared__ float vec1[64];
    __shared__ float vec2[64];
    __shared__ float mvals[64];

    const int n   = blockIdx.x;
    const int tid = threadIdx.x;

    const float w00 = fcW[0], w01 = fcW[1], w10 = fcW[2], w11 = fcW[3];
    const float b0 = fcB[0], b1 = fcB[1];

    const float* x0p = x + (size_t)(n * 2 + 0) * 4096;
    const float* x1p = x + (size_t)(n * 2 + 1) * 4096;

    // Phase 0: fc(2->2) + ReLU, build h0 and h1^T in LDS (padded stride 65)
#pragma unroll
    for (int q = 0; q < 16; ++q) {
        int p = q * 256 + tid;
        int i = p >> 6, j = p & 63;
        float a0 = x0p[p], a1 = x1p[p];
        float h0v = fmaxf(__builtin_fmaf(a0, w00, __builtin_fmaf(a1, w01, b0)), 0.0f);
        float h1v = fmaxf(__builtin_fmaf(a0, w10, __builtin_fmaf(a1, w11, b1)), 0.0f);
        h0 [i * LDSTR + j] = h0v;
        h1t[j * LDSTR + i] = h1v;
    }
    __syncthreads();

    // vec1 = att(h0)                               (a = i, k = j)
    att_phase<false, false, false>(h0, nullptr, nullptr,
                                   p1W[0], p1W[1], p1b[0], p1s[0], p1t[0],
                                   mvals, vec1);
    // vec2 = att(h1^T scaled by vec1 along k=i)    (a = j, k = i)
    att_phase<false, true, false>(h1t, nullptr, vec1,
                                  p2W[0], p2W[1], p2b[0], p2s[0], p2t[0],
                                  mvals, vec2);
    // out  = att(h0 * vec1[rows] * vec2[cols]), is_last -> plain softmax
    att_phase<true, true, true>(h0, vec1, vec2,
                                oW[0], oW[1], ob[0], oS[0], oT[0],
                                mvals, out + n * 64);
}

extern "C" void kernel_launch(void* const* d_in, const int* in_sizes, int n_in,
                              void* d_out, int out_size, void* d_ws, size_t ws_size,
                              hipStream_t stream)
{
    const float* x   = (const float*)d_in[0];
    const float* fcW = (const float*)d_in[1];
    const float* fcB = (const float*)d_in[2];
    const float* p1W = (const float*)d_in[3];
    const float* p1b = (const float*)d_in[4];
    const float* p1s = (const float*)d_in[5];
    const float* p1t = (const float*)d_in[6];
    const float* p2W = (const float*)d_in[7];
    const float* p2b = (const float*)d_in[8];
    const float* p2s = (const float*)d_in[9];
    const float* p2t = (const float*)d_in[10];
    const float* oW  = (const float*)d_in[11];
    const float* ob  = (const float*)d_in[12];
    const float* oS  = (const float*)d_in[13];
    const float* oT  = (const float*)d_in[14];

    const int N = in_sizes[0] / (2 * 64 * 64);   // 128

    attnet_kernel<<<N, 256, 0, stream>>>(
        x, fcW, fcB, p1W, p1b, p1s, p1t, p2W, p2b, p2s, p2t,
        oW, ob, oS, oT, (float*)d_out);
}
